// LSTMCell_58385785422053
// MI455X (gfx1250) — matrix operand order repaired and checked by reading, not compiled, not run on hardware
//
#include <hip/hip_runtime.h>

typedef __attribute__((ext_vector_type(16))) __bf16 v16bf;
typedef __attribute__((ext_vector_type(8)))  __bf16 v8bf;
typedef __attribute__((ext_vector_type(8)))  float  v8f;
typedef __attribute__((ext_vector_type(4)))  int    v4i;

#define AS1 __attribute__((address_space(1)))
#define AS3 __attribute__((address_space(3)))

#define LSTM_BATCH 4096
#define LSTM_HID   2048
#define LSTM_K     4096      // combined = hidden + input
#define LSTM_4H    8192

#define BM 128               // block M tile
#define BJ 64                // block j tile (per gate; 4 gates staged)
#define BK 32                // K step (one bf16 WMMA)
#define SA 40                // padded LDS row stride in bf16 (80B -> conflict-free b128 reads)
#define NKSTEP (LSTM_K / BK)

#if __has_builtin(__builtin_amdgcn_global_load_async_to_lds_b128)
#define USE_ASYNC_LDS 1
#else
#define USE_ASYNC_LDS 0
#endif

__device__ __forceinline__ void wait_async0() {
#if __has_builtin(__builtin_amdgcn_s_wait_asynccnt)
    __builtin_amdgcn_s_wait_asynccnt(0);
#else
    asm volatile("s_wait_asynccnt 0x0" ::: "memory");
#endif
}

// ---------------------------------------------------------------------------
// fp32 -> bf16 conversion kernels (one streaming pass; ~15us at 23.3 TB/s).
// After this the bf16 working set (32MB act + 64MB W) fits in the 192MB L2.
// ---------------------------------------------------------------------------
__global__ void cvt_combined_kernel(const float* __restrict__ x_t,
                                    const float* __restrict__ h_prev,
                                    __bf16* __restrict__ comb) {
    long long t = (long long)blockIdx.x * blockDim.x + threadIdx.x;
    long long e = t * 4;
    if (e >= (long long)LSTM_BATCH * LSTM_K) return;
    int b = (int)(e >> 12);
    int c = (int)(e & (LSTM_K - 1));           // 4-chunk never straddles 2048
    const float* src = (c < LSTM_HID)
        ? (h_prev + (long long)b * LSTM_HID + c)
        : (x_t    + (long long)b * LSTM_HID + (c - LSTM_HID));
    float4 v = *(const float4*)src;
    union { __bf16 h[4]; unsigned long long u; } o;
    o.h[0] = (__bf16)v.x; o.h[1] = (__bf16)v.y;
    o.h[2] = (__bf16)v.z; o.h[3] = (__bf16)v.w;
    *(unsigned long long*)(comb + e) = o.u;
}

__global__ void cvt_weights_kernel(const float* __restrict__ W_f,
                                   const float* __restrict__ W_i,
                                   const float* __restrict__ W_c,
                                   const float* __restrict__ W_o,
                                   __bf16* __restrict__ wcat) {
    long long t = (long long)blockIdx.x * blockDim.x + threadIdx.x;
    long long e = t * 4;
    if (e >= (long long)LSTM_4H * LSTM_K) return;
    int n = (int)(e >> 12);
    int c = (int)(e & (LSTM_K - 1));
    int g = n >> 11;
    int j = n & (LSTM_HID - 1);
    const float* Wg = (g == 0) ? W_f : (g == 1) ? W_i : (g == 2) ? W_c : W_o;
    float4 v = *(const float4*)(Wg + (long long)j * LSTM_K + c);
    union { __bf16 h[4]; unsigned long long u; } o;
    o.h[0] = (__bf16)v.x; o.h[1] = (__bf16)v.y;
    o.h[2] = (__bf16)v.z; o.h[3] = (__bf16)v.w;
    *(unsigned long long*)(wcat + e) = o.u;
}

// ---------------------------------------------------------------------------
__device__ __forceinline__ v16bf pack16(v8bf lo, v8bf hi) {
    v16bf r;
#pragma unroll
    for (int i = 0; i < 8; ++i) { r[i] = lo[i]; r[i + 8] = hi[i]; }
    return r;
}

__device__ __forceinline__ float fsigmoid(float x) {
    return 1.0f / (1.0f + __expf(-x));
}
__device__ __forceinline__ float ftanh(float x) {
    return 1.0f - 2.0f / (__expf(2.0f * x) + 1.0f);
}

// ---------------------------------------------------------------------------
// Fused 4-gate GEMM + LSTM epilogue with LDS double buffering.
// Block = 256 threads = 8 waves arranged 2(M) x 4(j).
// Per wave: 64x16 tile, all 4 gates -> 16 accumulators, 16 WMMAs per K-step.
// LDS per buffer: A 128x32 (pad 40) + B 4x64x32 (pad 40) = 30KB; x2 = 60KB.
// Copies use GLOBAL_LOAD_ASYNC_TO_LDS_B128 (ASYNCcnt), bypassing VGPRs.
// ---------------------------------------------------------------------------
__global__ void __launch_bounds__(256)
lstm_gemm_wmma_kernel(const __bf16* __restrict__ comb,
                      const __bf16* __restrict__ wcat,
                      const float* __restrict__ c_prev,
                      const float* __restrict__ b_f,
                      const float* __restrict__ b_i,
                      const float* __restrict__ b_c,
                      const float* __restrict__ b_o,
                      float* __restrict__ out) {
    __shared__ __bf16 sA[2][BM * SA];          // 2 * 128*40*2B = 20KB
    __shared__ __bf16 sB[2][4 * BJ * SA];      // 2 * 256*40*2B = 40KB

    const int t    = threadIdx.x;
    const int lane = t & 31;
    const int wid  = t >> 5;
    const int wm   = wid & 1;                  // 2 waves along M
    const int wj   = wid >> 1;                 // 4 waves along j
    const int m0 = blockIdx.x * BM;
    const int j0 = blockIdx.y * BJ;

    v8f zero = {};
    v8f acc[4][4];                             // [gate][mfrag]
#pragma unroll
    for (int g = 0; g < 4; ++g)
#pragma unroll
        for (int f = 0; f < 4; ++f) acc[g][f] = zero;

    // per-lane fragment addressing (CDNA5 WMMA VGPR layouts, CDNA5 ISA 7.12.2)
    const int frow  = lane & 15;
    const int aksel = (lane >> 4) << 3;        // A: K 0..7 / 8..15 (+16 for hi)
    const int bkoff = (lane >> 4) << 3;        // B: K 0..7 / 8..15 (+16 for hi)

    // ---- tile copy: global (L2-resident bf16) -> LDS ----
    // A tile: 512 x 16B chunks, 2/thread. B tile: 1024 x 16B chunks, 4/thread.
#if USE_ASYNC_LDS
#define COPY_TILES(P, KOFF)                                                          \
    {                                                                                \
        _Pragma("unroll")                                                            \
        for (int i = 0; i < 2; ++i) {                                                \
            int c = t + 256 * i, row = c >> 2, seg = c & 3;                          \
            const __bf16* g = comb + (size_t)(m0 + row) * LSTM_K + (KOFF) + seg * 8; \
            __bf16* l = &sA[P][row * SA + seg * 8];                                  \
            __builtin_amdgcn_global_load_async_to_lds_b128(                          \
                (AS1 v4i*)g, (AS3 v4i*)l, 0, 0);                                     \
        }                                                                            \
        _Pragma("unroll")                                                            \
        for (int i = 0; i < 4; ++i) {                                                \
            int c = t + 256 * i, grow = c >> 2, seg = c & 3;                         \
            int gt = grow >> 6, jr = grow & 63;                                      \
            const __bf16* g = wcat +                                                 \
                (size_t)(gt * LSTM_HID + j0 + jr) * LSTM_K + (KOFF) + seg * 8;       \
            __bf16* l = &sB[P][grow * SA + seg * 8];                                 \
            __builtin_amdgcn_global_load_async_to_lds_b128(                          \
                (AS1 v4i*)g, (AS3 v4i*)l, 0, 0);                                     \
        }                                                                            \
    }
#else
    v8bf ra[2], rb[4];
#define FETCH_TILES(KOFF)                                                            \
    {                                                                                \
        _Pragma("unroll")                                                            \
        for (int i = 0; i < 2; ++i) {                                                \
            int c = t + 256 * i, row = c >> 2, seg = c & 3;                          \
            ra[i] = *(const v8bf*)(comb + (size_t)(m0 + row) * LSTM_K + (KOFF) + seg * 8); \
        }                                                                            \
        _Pragma("unroll")                                                            \
        for (int i = 0; i < 4; ++i) {                                                \
            int c = t + 256 * i, grow = c >> 2, seg = c & 3;                         \
            int gt = grow >> 6, jr = grow & 63;                                      \
            rb[i] = *(const v8bf*)(wcat +                                            \
                (size_t)(gt * LSTM_HID + j0 + jr) * LSTM_K + (KOFF) + seg * 8);      \
        }                                                                            \
    }
#define STORE_TILES(P)                                                               \
    {                                                                                \
        _Pragma("unroll")                                                            \
        for (int i = 0; i < 2; ++i) {                                                \
            int c = t + 256 * i, row = c >> 2, seg = c & 3;                          \
            *(v8bf*)(&sA[P][row * SA + seg * 8]) = ra[i];                            \
        }                                                                            \
        _Pragma("unroll")                                                            \
        for (int i = 0; i < 4; ++i) {                                                \
            int c = t + 256 * i, grow = c >> 2, seg = c & 3;                         \
            *(v8bf*)(&sB[P][grow * SA + seg * 8]) = rb[i];                           \
        }                                                                            \
    }
#endif

    // ---- compute one K-step from LDS buffer P ----
#define COMPUTE_STEP(P)                                                              \
    {                                                                                \
        v16bf afrag[4];                                                              \
        _Pragma("unroll")                                                            \
        for (int f = 0; f < 4; ++f) {                                                \
            const __bf16* q = &sA[P][(wm * 64 + f * 16 + frow) * SA + aksel];        \
            afrag[f] = pack16(*(const v8bf*)q, *(const v8bf*)(q + 16));              \
        }                                                                            \
        _Pragma("unroll")                                                            \
        for (int g = 0; g < 4; ++g) {                                                \
            const __bf16* q = &sB[P][(g * BJ + wj * 16 + frow) * SA + bkoff];        \
            v16bf bfrag = pack16(*(const v8bf*)q, *(const v8bf*)(q + 16));           \
            _Pragma("unroll")                                                        \
            for (int f = 0; f < 4; ++f)                                              \
                acc[g][f] = __builtin_amdgcn_wmma_f32_16x16x32_bf16(                 \
                    false, afrag[f], false, bfrag, (short)0, acc[g][f],              \
                    false, false);                                                   \
        }                                                                            \
    }

#if USE_ASYNC_LDS
    COPY_TILES(0, 0)
    wait_async0();
    __syncthreads();
    for (int kt = 0; kt < NKSTEP; ++kt) {
        int p = kt & 1;
        if (kt + 1 < NKSTEP) COPY_TILES(p ^ 1, (kt + 1) * BK)
        COMPUTE_STEP(p)
        wait_async0();       // this wave's async copies into p^1 landed
        __syncthreads();     // everyone done reading p / writing p^1
    }
#else
    FETCH_TILES(0)
    for (int kt = 0; kt < NKSTEP; ++kt) {
        int p = kt & 1;
        STORE_TILES(p)
        __syncthreads();
        if (kt + 1 < NKSTEP) FETCH_TILES((kt + 1) * BK)
        COMPUTE_STEP(p)
        __syncthreads();
    }
#endif

    // ---- fused LSTM epilogue ----
    // C/D layout: VGPR e, lanes 0-15 -> M=e, N=lane; lanes 16-31 -> M=e+8.
    const int ncol  = lane & 15;
    const int mbase = (lane >> 4) << 3;
    const int j = j0 + wj * 16 + ncol;
    const float bfv = b_f[j], biv = b_i[j], bcv = b_c[j], bov = b_o[j];

#pragma unroll
    for (int f = 0; f < 4; ++f) {
#pragma unroll
        for (int e = 0; e < 8; ++e) {
            const int m = m0 + wm * 64 + f * 16 + mbase + e;
            float ft = fsigmoid(acc[0][f][e] + bfv);
            float it = fsigmoid(acc[1][f][e] + biv);
            float ct = ftanh   (acc[2][f][e] + bcv);
            float ot = fsigmoid(acc[3][f][e] + bov);
            float cp = c_prev[(size_t)m * LSTM_HID + j];
            float cn = ft * cp + it * ct;
            out[(size_t)m * LSTM_HID + j] = ot * ftanh(cn);                           // h_t
            out[(size_t)LSTM_BATCH * LSTM_HID + (size_t)m * LSTM_HID + j] = cn;       // c_t
        }
    }
}

// ---------------------------------------------------------------------------
extern "C" void kernel_launch(void* const* d_in, const int* in_sizes, int n_in,
                              void* d_out, int out_size, void* d_ws, size_t ws_size,
                              hipStream_t stream) {
    const float* x_t    = (const float*)d_in[0];
    const float* h_prev = (const float*)d_in[1];
    const float* c_prev = (const float*)d_in[2];
    const float* W_f    = (const float*)d_in[3];
    const float* b_f    = (const float*)d_in[4];
    const float* W_i    = (const float*)d_in[5];
    const float* b_i    = (const float*)d_in[6];
    const float* W_c    = (const float*)d_in[7];
    const float* b_c    = (const float*)d_in[8];
    const float* W_o    = (const float*)d_in[9];
    const float* b_o    = (const float*)d_in[10];

    __bf16* comb = (__bf16*)d_ws;                              // [4096, 4096] bf16 (32MB)
    __bf16* wcat = comb + (long long)LSTM_BATCH * LSTM_K;      // [8192, 4096] bf16 (64MB)

    {
        long long n = (long long)LSTM_BATCH * LSTM_K / 4;
        cvt_combined_kernel<<<(unsigned)((n + 255) / 256), 256, 0, stream>>>(x_t, h_prev, comb);
    }
    {
        long long n = (long long)LSTM_4H * LSTM_K / 4;
        cvt_weights_kernel<<<(unsigned)((n + 255) / 256), 256, 0, stream>>>(W_f, W_i, W_c, W_o, wcat);
    }
    dim3 grid(LSTM_BATCH / BM, LSTM_HID / BJ);
    lstm_gemm_wmma_kernel<<<grid, 256, 0, stream>>>(comb, wcat, c_prev,
                                                    b_f, b_i, b_c, b_o,
                                                    (float*)d_out);
}
